// GNNEncoder_4964982194350
// MI455X (gfx1250) — matrix operand order repaired and checked by reading, not compiled, not run on hardware
//
#include <hip/hip_runtime.h>
#include <hip/hip_bf16.h>

// GraphSAGE 3-layer for MI455X (gfx1250, wave32, WMMA bf16).
// v4: f32->bf16x2 pack via single v_perm_b32 (+round bias), cutting A-staging VALU ~3x.
//     Fallback chain: v_cvt_pk_bf16_f32 -> v_perm_b32 -> scalar RNE, all __has_builtin-guarded.

typedef __attribute__((ext_vector_type(16))) __bf16 v16bf;
typedef __attribute__((ext_vector_type(2)))  __bf16 v2bf;
typedef __attribute__((ext_vector_type(8)))  float  v8f;

#define NNODES 50000
#define NEDGES 800000
#define DH     512
#define LDSTR  36   // LDS row stride in ushorts (32 data + 4 pad -> 72B, kills bank conflicts)

union Frag16 { v16bf v; unsigned long long q[4]; };

__device__ __forceinline__ unsigned short f2bf(float f) {
  // round-to-nearest-even f32 -> bf16 (scalar fallback)
  unsigned int u = __builtin_bit_cast(unsigned int, f);
  u += 0x7FFFu + ((u >> 16) & 1u);
  return (unsigned short)(u >> 16);
}

__device__ __forceinline__ unsigned long long pack4bf(float4 v) {
#if __has_builtin(__builtin_amdgcn_cvt_pk_bf16_f32)
  v2bf lo = __builtin_amdgcn_cvt_pk_bf16_f32(v.x, v.y);
  v2bf hi = __builtin_amdgcn_cvt_pk_bf16_f32(v.z, v.w);
  return (unsigned long long)__builtin_bit_cast(unsigned int, lo)
       | ((unsigned long long)__builtin_bit_cast(unsigned int, hi) << 32);
#elif __has_builtin(__builtin_amdgcn_perm)
  // round-to-nearest (ties away) then take high halves of two f32 with one v_perm_b32.
  unsigned int ux = __builtin_bit_cast(unsigned int, v.x) + 0x8000u;
  unsigned int uy = __builtin_bit_cast(unsigned int, v.y) + 0x8000u;
  unsigned int uz = __builtin_bit_cast(unsigned int, v.z) + 0x8000u;
  unsigned int uw = __builtin_bit_cast(unsigned int, v.w) + 0x8000u;
  // perm(src0, src1, sel): sel bytes 0-3 -> src1, 4-7 -> src0
  unsigned int lo = __builtin_amdgcn_perm(uy, ux, 0x07060302u); // {uy.hi16, ux.hi16}
  unsigned int hi = __builtin_amdgcn_perm(uw, uz, 0x07060302u); // {uw.hi16, uz.hi16}
  return (unsigned long long)lo | ((unsigned long long)hi << 32);
#else
  return (unsigned long long)f2bf(v.x)
       | ((unsigned long long)f2bf(v.y) << 16)
       | ((unsigned long long)f2bf(v.z) << 32)
       | ((unsigned long long)f2bf(v.w) << 48);
#endif
}

__global__ __launch_bounds__(256) void deg_count(const long long* __restrict__ dstI,
                                                 float* __restrict__ deg, int nE) {
  int i = blockIdx.x * 256 + threadIdx.x;
  if (i < nE) atomicAdd(&deg[(int)dstI[i]], 1.0f);
}

__global__ __launch_bounds__(256) void inv_deg_k(const float* __restrict__ deg,
                                                 float* __restrict__ inv, int n) {
  int i = blockIdx.x * 256 + threadIdx.x;
  if (i < n) inv[i] = 1.0f / fmaxf(deg[i], 1.0f);
}

// Pack [W_l | W_r] rows into bf16 [N][1024]; B-tile for out = A @ W.T is just W's rows.
__global__ __launch_bounds__(256) void pack_w(const float* __restrict__ Wl,
                                              const float* __restrict__ Wr,
                                              unsigned short* __restrict__ Wb, int n) {
  int i = blockIdx.x * 256 + threadIdx.x;
  if (i >= (n << 10)) return;
  int row = i >> 10, k = i & 1023;
  float v = (k < DH) ? Wl[row * DH + k] : Wr[row * DH + (k - DH)];
  Wb[i] = f2bf(v);
}

// 128 threads per edge: coalesced float4 read of source row, 4 f32 atomics into msg_sum.
// msg_sum (102 MB) is L2-resident on MI455X (192 MB L2) -> atomics resolve at L2 rate.
__global__ __launch_bounds__(256) void scatter_add(const float* __restrict__ feat,
                                                   const long long* __restrict__ srcI,
                                                   const long long* __restrict__ dstI,
                                                   float* __restrict__ msg, int nE) {
  int e = blockIdx.x * 2 + (threadIdx.x >> 7);
  if (e >= nE) return;
  int t = (threadIdx.x & 127) << 2;
  int s = (int)srcI[e], d = (int)dstI[e];
  const float4 v = *(const float4*)(feat + (size_t)s * DH + t);
  float* o = msg + (size_t)d * DH + t;
  atomicAdd(o, v.x); atomicAdd(o + 1, v.y); atomicAdd(o + 2, v.z); atomicAdd(o + 3, v.w);
}

// Fused SAGE layer GEMM: out[r][n] = relu?( sum_k A[r][k]*Wb[n][k] + bias[n] )
// A[r][k] = (k<512) ? msg[r][k]*invdeg[r] : xin[r][k-512]   (K = 1024 fixed)
// 128x128 tile, 8 waves (4x2), each wave: 2x4 accumulators of v_wmma_f32_16x16x32_bf16.
// LDS double-buffered: stage tile k+1 while computing tile k; one barrier per k-step.
__global__ __launch_bounds__(256)
void sage_gemm(const float* __restrict__ msg, const float* __restrict__ invdeg,
               const float* __restrict__ xin, const unsigned short* __restrict__ Wb,
               const float* __restrict__ bias, float* __restrict__ out,
               int M, int N, int doRelu) {
  __shared__ unsigned short As[2][128 * LDSTR];
  __shared__ unsigned short Bs[2][128 * LDSTR];

  const int tid  = threadIdx.x;
  const int lane = tid & 31;
  const int wave = tid >> 5;
  const int wm   = wave & 3;    // 32-row chunk within tile
  const int wn   = wave >> 2;   // 64-col chunk within tile
  const int half = lane >> 4;   // ISA: lanes 16..31 hold K 8..15 and 24..31
  const int l16  = lane & 15;

  const int rowBlk = blockIdx.x * 128;
  const int colBlk = blockIdx.y * 128;

  v8f az;
  #pragma unroll
  for (int i = 0; i < 8; ++i) az[i] = 0.0f;
  v8f acc[2][4];
  #pragma unroll
  for (int i = 0; i < 2; ++i)
    #pragma unroll
    for (int j = 0; j < 4; ++j) acc[i][j] = az;

  // ---- staging geometry: A = 32 rows/pass x 8 thr/row (4 f32); B = 64 rows/pass x 4 thr/row ----
  const int arow = tid >> 3;
  const int acol = (tid & 7) << 2;
  const int brow = tid >> 2;
  const int bcol = (tid & 3) << 3;

  // Hoisted, clamped per-row pointers + mean scale. Rows >= M clamp to M-1: D-row m only
  // depends on A-row m, so garbage in pad rows never reaches valid outputs (stores guarded).
  const float* aMsg[4];
  const float* aX[4];
  float aScale[4];
  #pragma unroll
  for (int p = 0; p < 4; ++p) {
    int gr = rowBlk + p * 32 + arow;
    if (gr > M - 1) gr = M - 1;
    aMsg[p]   = msg + (size_t)gr * DH + acol;
    aX[p]     = xin + (size_t)gr * DH + acol;
    aScale[p] = invdeg[gr];
  }
  const unsigned short* bSrc[2];
  #pragma unroll
  for (int p = 0; p < 2; ++p) {
    int gn = colBlk + p * 64 + brow;     // N is a multiple of 128: always in range
    bSrc[p] = Wb + (size_t)gn * 1024 + bcol;
  }

  auto stage = [&](int buf, int k0) {
    unsigned short* Asb = As[buf];
    unsigned short* Bsb = Bs[buf];
    const bool msgPhase = k0 < DH;       // uniform: k0 is scalar, 512 % 32 == 0
    const int  kk = msgPhase ? k0 : (k0 - DH);
    #pragma unroll
    for (int p = 0; p < 4; ++p) {
      float4 v = *(const float4*)((msgPhase ? aMsg[p] : aX[p]) + kk);
      if (msgPhase) {
        float s = aScale[p];
        v.x *= s; v.y *= s; v.z *= s; v.w *= s;
      }
      *(unsigned long long*)(&Asb[(p * 32 + arow) * LDSTR + acol]) = pack4bf(v);
    }
    #pragma unroll
    for (int p = 0; p < 2; ++p) {
      const unsigned long long* s = (const unsigned long long*)(bSrc[p] + k0);
      unsigned long long q0 = s[0], q1 = s[1];
      unsigned long long* d = (unsigned long long*)(&Bsb[(p * 64 + brow) * LDSTR + bcol]);
      d[0] = q0; d[1] = q1;
    }
  };

  auto compute = [&](int buf) {
    const unsigned short* Asb = As[buf];
    const unsigned short* Bsb = Bs[buf];
    Frag16 a[2], b[4];
    #pragma unroll
    for (int mi = 0; mi < 2; ++mi) {
      int base = (wm * 32 + mi * 16 + l16) * LDSTR + half * 8;
      // A 16-bit 16x32: VGPR 0..3 -> K = half*8 + 0..7; VGPR 4..7 -> K = 16 + half*8 + 0..7
      a[mi].q[0] = *(const unsigned long long*)(&Asb[base]);
      a[mi].q[1] = *(const unsigned long long*)(&Asb[base + 4]);
      a[mi].q[2] = *(const unsigned long long*)(&Asb[base + 16]);
      a[mi].q[3] = *(const unsigned long long*)(&Asb[base + 20]);
    }
    #pragma unroll
    for (int ni = 0; ni < 4; ++ni) {
      int base = (wn * 64 + ni * 16 + l16) * LDSTR + half * 8;
      // B 16-bit 32x16: lane = column, elements 0..7 -> K = half*8 + 0..7; 8..15 -> K = 16 + half*8 + 0..7
      b[ni].q[0] = *(const unsigned long long*)(&Bsb[base]);
      b[ni].q[1] = *(const unsigned long long*)(&Bsb[base + 4]);
      b[ni].q[2] = *(const unsigned long long*)(&Bsb[base + 16]);
      b[ni].q[3] = *(const unsigned long long*)(&Bsb[base + 20]);
    }
    #pragma unroll
    for (int mi = 0; mi < 2; ++mi)
      #pragma unroll
      for (int ni = 0; ni < 4; ++ni)
        acc[mi][ni] = __builtin_amdgcn_wmma_f32_16x16x32_bf16(
            false, a[mi].v, false, b[ni].v, (short)0, acc[mi][ni], false, false);
  };

  // ---- pipelined main loop: one barrier per k-step ----
  stage(0, 0);
  __syncthreads();
  int buf = 0;
  #pragma unroll 2
  for (int k0 = 0; k0 < 1024 - 32; k0 += 32) {
    stage(buf ^ 1, k0 + 32);   // prefetch next tile into other buffer
    compute(buf);              // WMMA on current tile
    __syncthreads();           // staging visible; current buffer free for reuse
    buf ^= 1;
  }
  compute(buf);

  // ---- epilogue: C layout -> lane 0-15: M=v, lane 16-31: M=v+8; N = lane%16 ----
  #pragma unroll
  for (int mi = 0; mi < 2; ++mi) {
    int rbase = rowBlk + wm * 32 + mi * 16 + half * 8;
    #pragma unroll
    for (int ni = 0; ni < 4; ++ni) {
      int col = colBlk + wn * 64 + ni * 16 + l16;
      float bv = bias[col];
      #pragma unroll
      for (int v = 0; v < 8; ++v) {
        int r = rbase + v;
        if (r < M) {
          float o = acc[mi][ni][v] + bv;
          if (doRelu) o = fmaxf(o, 0.0f);
          out[(size_t)r * N + col] = o;
        }
      }
    }
  }
}

extern "C" void kernel_launch(void* const* d_in, const int* in_sizes, int n_in,
                              void* d_out, int out_size, void* d_ws, size_t ws_size,
                              hipStream_t stream) {
  (void)in_sizes; (void)n_in; (void)out_size; (void)ws_size;
  const float*     x    = (const float*)d_in[0];
  const long long* eidx = (const long long*)d_in[1];   // int64 [2, 800000]
  const float* W1l = (const float*)d_in[2];
  const float* b1  = (const float*)d_in[3];
  const float* W1r = (const float*)d_in[4];
  const float* W2l = (const float*)d_in[5];
  const float* b2  = (const float*)d_in[6];
  const float* W2r = (const float*)d_in[7];
  const float* W3l = (const float*)d_in[8];
  const float* b3  = (const float*)d_in[9];
  const float* W3r = (const float*)d_in[10];

  const long long* srcI = eidx;
  const long long* dstI = eidx + NEDGES;

  char* ws = (char*)d_ws;
  size_t off = 0;
  auto take = [&](size_t bytes) -> char* {
    char* p = ws + off;
    off = (off + bytes + 255) & ~(size_t)255;
    return p;
  };
  float* deg  = (float*)take((size_t)NNODES * 4);
  float* inv  = (float*)take((size_t)NNODES * 4);
  float* msgb = (float*)take((size_t)NNODES * DH * 4);
  float* h1   = (float*)take((size_t)NNODES * DH * 4);
  float* h2   = (float*)take((size_t)NNODES * DH * 4);
  unsigned short* Wb1 = (unsigned short*)take((size_t)512 * 1024 * 2);
  unsigned short* Wb2 = (unsigned short*)take((size_t)512 * 1024 * 2);
  unsigned short* Wb3 = (unsigned short*)take((size_t)256 * 1024 * 2);

  const size_t msgBytes = (size_t)NNODES * DH * 4;
  const int scatterBlocks = (NEDGES + 1) / 2;

  // degree (layer-invariant) + weight packs
  hipMemsetAsync(deg, 0, (size_t)NNODES * 4, stream);
  deg_count<<<(NEDGES + 255) / 256, 256, 0, stream>>>(dstI, deg, NEDGES);
  inv_deg_k<<<(NNODES + 255) / 256, 256, 0, stream>>>(deg, inv, NNODES);
  pack_w<<<(512 * 1024 + 255) / 256, 256, 0, stream>>>(W1l, W1r, Wb1, 512);
  pack_w<<<(512 * 1024 + 255) / 256, 256, 0, stream>>>(W2l, W2r, Wb2, 512);
  pack_w<<<(256 * 1024 + 255) / 256, 256, 0, stream>>>(W3l, W3r, Wb3, 256);

  dim3 g512((NNODES + 127) / 128, 4);
  dim3 g256((NNODES + 127) / 128, 2);

  // layer 1
  hipMemsetAsync(msgb, 0, msgBytes, stream);
  scatter_add<<<scatterBlocks, 256, 0, stream>>>(x, srcI, dstI, msgb, NEDGES);
  sage_gemm<<<g512, 256, 0, stream>>>(msgb, inv, x, Wb1, b1, h1, NNODES, 512, 1);
  // layer 2
  hipMemsetAsync(msgb, 0, msgBytes, stream);
  scatter_add<<<scatterBlocks, 256, 0, stream>>>(h1, srcI, dstI, msgb, NEDGES);
  sage_gemm<<<g512, 256, 0, stream>>>(msgb, inv, h1, Wb2, b2, h2, NNODES, 512, 1);
  // layer 3 (no relu) -> d_out
  hipMemsetAsync(msgb, 0, msgBytes, stream);
  scatter_add<<<scatterBlocks, 256, 0, stream>>>(h2, srcI, dstI, msgb, NEDGES);
  sage_gemm<<<g256, 256, 0, stream>>>(msgb, inv, h2, Wb3, b3, (float*)d_out, NNODES, 256, 0);
}
